// GPTNeoSelfAttention_88527865905620
// MI455X (gfx1250) — matrix operand order repaired and checked by reading, not compiled, not run on hardware
//
#include <hip/hip_runtime.h>
#include <hip/hip_bf16.h>
#include <stdint.h>

// ---------------- problem constants ----------------
#define Bsz 2
#define Sq  2048
#define Dm  768
#define Hh  12
#define HDh 64

// ---------------- vector types ----------------
typedef __attribute__((ext_vector_type(16))) _Float16 v16h;
typedef __attribute__((ext_vector_type(8)))  _Float16 v8h;
typedef __attribute__((ext_vector_type(8)))  float    v8f;
typedef __attribute__((ext_vector_type(4)))  float    v4f;
typedef __attribute__((ext_vector_type(4)))  unsigned int v4u;
typedef __attribute__((ext_vector_type(8)))  int      v8i;
typedef __attribute__((ext_vector_type(4)))  int      v4i;

// ---------------- WMMA wrapper ----------------
__device__ __forceinline__ v8f wmma_f16(v16h a, v16h b, v8f c) {
  return __builtin_amdgcn_wmma_f32_16x16x32_f16(false, a, false, b, (short)0, c,
                                                false, false);
}

// ---------------- operand builders ----------------
// A operand (16x32 f16, M x K): element i of lane l holds
//   M = l%16, K = 16*(i/8) + 8*(l/16) + i%8
// -> two contiguous 8-half runs at K = hi8 and 16+hi8, hi8 = 8*(l>=16)
// B operand (32x16 f16, K x N): element i of lane l holds
//   N = l%16, K = 16*(i/8) + 8*(l/16) + i%8
// -> two contiguous 8-half runs at K = hi16/2 and 16+hi16/2, hi16 = 16*(l>=16)
__device__ __forceinline__ v8h ld8h(const _Float16* p) {
  return *(const v8h*)p;
}
__device__ __forceinline__ v16h mk16(v8h a, v8h b) {
  v16h r;
#pragma unroll
  for (int i = 0; i < 8; ++i) { r[i] = a[i]; r[i + 8] = b[i]; }
  return r;
}
__device__ __forceinline__ v16h a_f16(const _Float16* rowbase, int hi8) {
  return mk16(ld8h(rowbase + hi8), ld8h(rowbase + 16 + hi8));
}
__device__ __forceinline__ v16h b_f16(const _Float16* rowbase, int hi16) {
  const _Float16* p = rowbase + (hi16 >> 1);
  return mk16(ld8h(p), ld8h(p + 16));
}

// ---------------- TDM: 2D tensor tile -> LDS ----------------
// D# per CDNA5 ISA ch.8.  Pad: 16B per 128B row -> LDS row stride 72 halves.
//   pad_interval code 4 = 32 DWORDS (128B), pad_amount code 3 = 4 DWORDS (16B)
__device__ __forceinline__ void tdm_load_2d(uint32_t lds_addr, const void* gptr,
                                            uint32_t rowlen, uint32_t nrows,
                                            uint64_t row_stride) {
  uint64_t ga = (uint64_t)(uintptr_t)gptr;
  v4u g0;
  g0[0] = 1u;                                            // count=1, user desc
  g0[1] = lds_addr;                                      // LDS byte address
  g0[2] = (uint32_t)ga;                                  // global addr lo
  g0[3] = (uint32_t)((ga >> 32) & 0x1FFFFFFu) | (2u << 30);  // addr hi | type=2
  v8i g1;
  g1[0] = (int)((1u << 16) | (1u << 20) | (4u << 22) | (3u << 25));
  //            data_size=2B  pad_en      interval=32DW  amount=4DW
  g1[1] = (int)((rowlen & 0xFFFFu) << 16);               // abar=0 | tdim0.lo
  g1[2] = (int)(((rowlen >> 16) & 0xFFFFu) | ((nrows & 0xFFFFu) << 16));
  g1[3] = (int)(((nrows >> 16) & 0xFFFFu) | ((rowlen & 0xFFFFu) << 16)); // tile_dim0
  g1[4] = (int)(nrows & 0xFFFFu);                        // tile_dim1 | tile_dim2=0
  g1[5] = (int)(uint32_t)(row_stride & 0xFFFFFFFFu);     // dim0_stride lo
  g1[6] = (int)(uint32_t)((row_stride >> 32) & 0xFFFFu); // dim0_stride hi
  g1[7] = 0;
  v4i z4 = {0, 0, 0, 0};
  v8i z8 = {0, 0, 0, 0, 0, 0, 0, 0};
  __builtin_amdgcn_tensor_load_to_lds(g0, g1, z4, z4, z8, 0);
}

// =====================================================================
// Kernel 0: fp32 -> f16 convert (one pass; feeds all GEMMs with pure f16)
// =====================================================================
__global__ __launch_bounds__(256) void cvt_kernel(const float* __restrict__ in,
                                                  _Float16* __restrict__ out) {
  const size_t i = ((size_t)blockIdx.x * 256 + threadIdx.x) * 8;
  v4f a = *(const v4f*)(in + i);
  v4f b = *(const v4f*)(in + i + 4);
  v8h r;
  r[0] = (_Float16)a.x; r[1] = (_Float16)a.y; r[2] = (_Float16)a.z; r[3] = (_Float16)a.w;
  r[4] = (_Float16)b.x; r[5] = (_Float16)b.y; r[6] = (_Float16)b.z; r[7] = (_Float16)b.w;
  *(v8h*)(out + i) = r;
}

// =====================================================================
// Kernel 1: fused QKV projection.  out = x @ W^T  (C[m,n] = sum_k x[m,k]W[n,k])
// grid = (Dm/64 = heads, B*S/128, 3 mats), block = 128 (4 waves)
// Wave owns 32 m-rows x 64 n-cols: 8 WMMA per 32-k step (B tiles reused 2x).
// Q,K -> f16 [b,h,s,hd];  V -> f16 transposed [b,h,hd,s]
// =====================================================================
__global__ __launch_bounds__(128) void qkv_kernel(
    const _Float16* __restrict__ xh, const _Float16* __restrict__ Wh3,
    _Float16* __restrict__ Qh, _Float16* __restrict__ Kh,
    _Float16* __restrict__ Vt) {
  const int lane = threadIdx.x & 31;
  const int wave = threadIdx.x >> 5;
  const int hi8  = (lane >> 4) << 3;
  const int hi16 = (lane >> 4) << 4;
  const int lrow = lane & 15;
  const int mat  = blockIdx.z;         // 0=Q 1=K 2=V
  const int h    = blockIdx.x;         // head index == 64-col tile
  const int m0   = blockIdx.y * 128 + wave * 32;
  const _Float16* W = Wh3 + (size_t)mat * Dm * Dm;
  const _Float16* arow0 = xh + (size_t)(m0 + lrow) * Dm;
  const _Float16* arow1 = xh + (size_t)(m0 + 16 + lrow) * Dm;

  v8f acc[2][4] = {};
  for (int kc = 0; kc < Dm; kc += 32) {
    v16h a0 = a_f16(arow0 + kc, hi8);
    v16h a1 = a_f16(arow1 + kc, hi8);
#pragma unroll
    for (int t = 0; t < 4; ++t) {
      v16h b = b_f16(W + (size_t)(h * 64 + t * 16 + lrow) * Dm + kc, hi16);
      acc[0][t] = wmma_f16(a0, b, acc[0][t]);
      acc[1][t] = wmma_f16(a1, b, acc[1][t]);
    }
  }
  _Float16* dstQK = (mat == 0) ? Qh : Kh;
#pragma unroll
  for (int mt = 0; mt < 2; ++mt) {
#pragma unroll
    for (int t = 0; t < 4; ++t) {
#pragma unroll
      for (int r = 0; r < 8; ++r) {
        int m = m0 + mt * 16 + r + hi8;     // flat token in [0, B*S)
        int bidx = m / Sq, s = m % Sq;
        int hd = t * 16 + lrow;
        _Float16 v = (_Float16)acc[mt][t][r];
        if (mat < 2)
          dstQK[((size_t)(bidx * Hh + h) * Sq + s) * HDh + hd] = v;
        else
          Vt[((size_t)(bidx * Hh + h) * HDh + hd) * Sq + s] = v;
      }
    }
  }
}

// =====================================================================
// Kernel 2: flash attention (causal + padding mask, NO 1/sqrt(d) scale).
// grid = (S/64, B*H), block = 128; wave w owns q rows [qb*64+16w, +16).
// K / V^T chunks (64 keys) are TDM-DMA'd into DOUBLE-BUFFERED padded LDS:
// chunk ci+1 streams in while chunk ci is computed (s_wait_tensorcnt 2).
// =====================================================================
#define KROW 72   // 64 + 8 halves pad per LDS row
#define KVBUF (2 * 64 * KROW)   // one K+V double-buffer slot (halves)
__global__ __launch_bounds__(128) void attn_kernel(
    const _Float16* __restrict__ Qh, const _Float16* __restrict__ Kh,
    const _Float16* __restrict__ Vt, const float* __restrict__ amask,
    _Float16* __restrict__ O) {
  __shared__ __align__(16) _Float16 smem[2 * KVBUF + 4 * 16 * 40];

  const int lane = threadIdx.x & 31;
  const int wave = threadIdx.x >> 5;
  const int hi8  = (lane >> 4) << 3;
  const int hi16 = (lane >> 4) << 4;
  const int lrow = lane & 15;
  _Float16* Psh = smem + 2 * KVBUF + wave * 16 * 40;

  const int qb = blockIdx.x;
  const int bh = blockIdx.y;
  const int b  = bh / Hh;
  const int h  = bh % Hh;
  const _Float16* Qbase = Qh + (size_t)bh * Sq * HDh;
  const _Float16* Kbase = Kh + (size_t)bh * Sq * HDh;
  const _Float16* Vbase = Vt + (size_t)bh * HDh * Sq;   // [HD][S]
  const int q0 = qb * 64 + wave * 16;

  // Q tile A-operands, resident for the whole kernel (d = 0..31, 32..63)
  const _Float16* qrow = Qbase + (size_t)(q0 + lrow) * HDh;
  v16h qa0 = a_f16(qrow, hi8);
  v16h qa1 = a_f16(qrow + 32, hi8);

  v8f oacc[4] = {};
  float mrow[8], srow[8];
#pragma unroll
  for (int r = 0; r < 8; ++r) { mrow[r] = -1e30f; srow[r] = 0.f; }

  const uint32_t ldsBase = (uint32_t)(uintptr_t)(void*)smem;
  const int nchunk = qb + 1;

  // prologue: issue chunk 0 into buffer 0
  if (wave == 0) {
    tdm_load_2d(ldsBase, Kbase, 64, 64, HDh);
    tdm_load_2d(ldsBase + 64 * KROW * 2, Vbase, 64, 64, Sq);
  }

  for (int ci = 0; ci < nchunk; ++ci) {
    const int kc = ci * 64;
    __syncthreads();                 // buffer (ci+1)&1 fully consumed (ci-1)
    if (wave == 0) {
      if (ci + 1 < nchunk) {
        const uint32_t nb = ldsBase + ((ci + 1) & 1) * KVBUF * 2;
        tdm_load_2d(nb, Kbase + (size_t)(kc + 64) * HDh, 64, 64, HDh);
        tdm_load_2d(nb + 64 * KROW * 2, Vbase + kc + 64, 64, 64, Sq);
        __builtin_amdgcn_s_wait_tensorcnt(2);  // chunk ci done; ci+1 in flight
      } else {
        __builtin_amdgcn_s_wait_tensorcnt(0);
      }
    }
    __syncthreads();                 // chunk ci visible to all waves

    const _Float16* Ksh = smem + (ci & 1) * KVBUF;
    const _Float16* Vsh = Ksh + 64 * KROW;

    // ---- scores: 4 tiles of 16 keys, K-dim = 64 via 2 chained WMMAs ----
    v8f sc[4];
#pragma unroll
    for (int t = 0; t < 4; ++t) {
      const _Float16* krow = Ksh + (t * 16 + lrow) * KROW;
      v8f s = {};
      s = wmma_f16(qa0, b_f16(krow, hi16), s);
      s = wmma_f16(qa1, b_f16(krow + 32, hi16), s);
      sc[t] = s;
    }

    // ---- causal + padding mask, row max (reduce over 16-lane half) ----
    int kcol[4]; bool mok[4];
#pragma unroll
    for (int t = 0; t < 4; ++t) {
      kcol[t] = kc + t * 16 + lrow;
      mok[t]  = (amask[b * Sq + kcol[t]] != 0.f);
    }
    float rmax[8];
#pragma unroll
    for (int r = 0; r < 8; ++r) {
      const int q = q0 + r + hi8;
      float mx = -1e30f;
#pragma unroll
      for (int t = 0; t < 4; ++t) {
        bool pass = (kcol[t] <= q) && mok[t];
        float v = pass ? sc[t][r] : -1e30f;
        sc[t][r] = v;
        mx = fmaxf(mx, v);
      }
      mx = fmaxf(mx, __shfl_xor(mx, 1, 32));
      mx = fmaxf(mx, __shfl_xor(mx, 2, 32));
      mx = fmaxf(mx, __shfl_xor(mx, 4, 32));
      mx = fmaxf(mx, __shfl_xor(mx, 8, 32));
      rmax[r] = mx;
    }

    // ---- online softmax update ----
#pragma unroll
    for (int r = 0; r < 8; ++r) {
      float mnew = fmaxf(mrow[r], rmax[r]);
      float corr = __expf(mrow[r] - mnew);
      mrow[r] = mnew;
      float rsum = 0.f;
#pragma unroll
      for (int t = 0; t < 4; ++t) {
        float p = (sc[t][r] <= -1e29f) ? 0.f : __expf(sc[t][r] - mnew);
        sc[t][r] = p;
        rsum += p;
      }
      rsum += __shfl_xor(rsum, 1, 32);
      rsum += __shfl_xor(rsum, 2, 32);
      rsum += __shfl_xor(rsum, 4, 32);
      rsum += __shfl_xor(rsum, 8, 32);
      srow[r] = srow[r] * corr + rsum;
#pragma unroll
      for (int d = 0; d < 4; ++d) oacc[d][r] *= corr;
    }

    // ---- P @ V : bounce P through LDS (C-layout -> A-layout) ----
#pragma unroll
    for (int half = 0; half < 2; ++half) {
#pragma unroll
      for (int tt = 0; tt < 2; ++tt) {
        const int t = half * 2 + tt;
#pragma unroll
        for (int r = 0; r < 8; ++r)
          Psh[(r + hi8) * 40 + tt * 16 + lrow] = (_Float16)sc[t][r];
      }
      // same-wave DS ops are in-order: store->load needs no explicit wait
      v16h pa = a_f16(Psh + lrow * 40, hi8);   // A: 16 q x 32 k
#pragma unroll
      for (int dt = 0; dt < 4; ++dt) {
        const _Float16* vrow = Vsh + (dt * 16 + lrow) * KROW + half * 32;
        oacc[dt] = wmma_f16(pa, b_f16(vrow, hi16), oacc[dt]);
      }
    }
  }

  // ---- normalize and write O (f16) in [b, s, h*64+hd] = [b,s,D] layout ----
  float inv[8];
#pragma unroll
  for (int r = 0; r < 8; ++r) inv[r] = 1.f / fmaxf(srow[r], 1e-20f);
#pragma unroll
  for (int dt = 0; dt < 4; ++dt) {
#pragma unroll
    for (int r = 0; r < 8; ++r) {
      int q = q0 + r + hi8;
      O[((size_t)b * Sq + q) * Dm + h * 64 + dt * 16 + lrow] =
          (_Float16)(oacc[dt][r] * inv[r]);
    }
  }
}

// =====================================================================
// Kernel 3: output projection  out = O @ Wo^T + bo   (fp32 out)
// grid = (Dm/64, B*S/128), block = 128; 32 m-rows x 64 n-cols per wave.
// =====================================================================
__global__ __launch_bounds__(128) void outproj_kernel(
    const _Float16* __restrict__ Oh, const _Float16* __restrict__ Woh,
    const float* __restrict__ bo, float* __restrict__ out) {
  const int lane = threadIdx.x & 31;
  const int wave = threadIdx.x >> 5;
  const int hi8  = (lane >> 4) << 3;
  const int hi16 = (lane >> 4) << 4;
  const int lrow = lane & 15;
  const int n0 = blockIdx.x * 64;
  const int m0 = blockIdx.y * 128 + wave * 32;
  const _Float16* arow0 = Oh + (size_t)(m0 + lrow) * Dm;
  const _Float16* arow1 = Oh + (size_t)(m0 + 16 + lrow) * Dm;

  v8f acc[2][4] = {};
  for (int kc = 0; kc < Dm; kc += 32) {
    v16h a0 = a_f16(arow0 + kc, hi8);
    v16h a1 = a_f16(arow1 + kc, hi8);
#pragma unroll
    for (int t = 0; t < 4; ++t) {
      v16h b = b_f16(Woh + (size_t)(n0 + t * 16 + lrow) * Dm + kc, hi16);
      acc[0][t] = wmma_f16(a0, b, acc[0][t]);
      acc[1][t] = wmma_f16(a1, b, acc[1][t]);
    }
  }
#pragma unroll
  for (int t = 0; t < 4; ++t) {
    const int n = n0 + t * 16 + lrow;
    const float bias = bo[n];
#pragma unroll
    for (int mt = 0; mt < 2; ++mt)
#pragma unroll
      for (int r = 0; r < 8; ++r)
        out[(size_t)(m0 + mt * 16 + r + hi8) * Dm + n] = acc[mt][t][r] + bias;
  }
}

// =====================================================================
// launch
// =====================================================================
extern "C" void kernel_launch(void* const* d_in, const int* in_sizes, int n_in,
                              void* d_out, int out_size, void* d_ws,
                              size_t ws_size, hipStream_t stream) {
  const float* x     = (const float*)d_in[0];
  const float* amask = (const float*)d_in[1];
  const float* Wq    = (const float*)d_in[2];
  const float* Wk    = (const float*)d_in[3];
  const float* Wv    = (const float*)d_in[4];
  const float* Wo    = (const float*)d_in[5];
  const float* bo    = (const float*)d_in[6];

  const size_t X_ELT = (size_t)Bsz * Sq * Dm;   // 3,145,728
  const size_t W_ELT = (size_t)Dm * Dm;         // 589,824
  const size_t QK_BYTES = (size_t)Bsz * Hh * Sq * HDh * sizeof(_Float16);

  char* ws = (char*)d_ws;
  _Float16* xh  = (_Float16*)(ws);
  _Float16* Wh3 = (_Float16*)(ws + X_ELT * 2);              // Wq,Wk,Wv f16
  _Float16* Woh = Wh3 + 3 * W_ELT;
  char* p = (char*)(Woh + W_ELT);
  _Float16* Qh = (_Float16*)(p);
  _Float16* Kh = (_Float16*)(p + QK_BYTES);
  _Float16* Vt = (_Float16*)(p + 2 * QK_BYTES);
  _Float16* Oh = (_Float16*)(p + 3 * QK_BYTES);

  // 0) fp32 -> f16 converts (x, Wq, Wk, Wv, Wo)
  cvt_kernel<<<dim3(X_ELT / (256 * 8)), 256, 0, stream>>>(x, xh);
  cvt_kernel<<<dim3(W_ELT / (256 * 8)), 256, 0, stream>>>(Wq, Wh3);
  cvt_kernel<<<dim3(W_ELT / (256 * 8)), 256, 0, stream>>>(Wk, Wh3 + W_ELT);
  cvt_kernel<<<dim3(W_ELT / (256 * 8)), 256, 0, stream>>>(Wv, Wh3 + 2 * W_ELT);
  cvt_kernel<<<dim3(W_ELT / (256 * 8)), 256, 0, stream>>>(Wo, Woh);

  dim3 blk(128);
  qkv_kernel<<<dim3(Dm / 64, (Bsz * Sq) / 128, 3), blk, 0, stream>>>(
      xh, Wh3, Qh, Kh, Vt);
  attn_kernel<<<dim3(Sq / 64, Bsz * Hh), blk, 0, stream>>>(
      Qh, Kh, Vt, amask, Oh);
  outproj_kernel<<<dim3(Dm / 64, (Bsz * Sq) / 128), blk, 0, stream>>>(
      Oh, Woh, bo, (float*)d_out);
}
